// window_attention_38379827757319
// MI455X (gfx1250) — matrix operand order repaired and checked by reading, not compiled, not run on hardware
//
#include <hip/hip_runtime.h>

typedef __bf16 bf16_t;
typedef __attribute__((ext_vector_type(16))) __bf16 v16bf;
typedef __attribute__((ext_vector_type(8)))  __bf16 v8bf;
typedef __attribute__((ext_vector_type(8)))  float  v8f;
typedef __attribute__((ext_vector_type(4)))  float  v4f;

#define WMMA_BF16(a, b, c) \
  __builtin_amdgcn_wmma_f32_16x16x32_bf16(false, (a), false, (b), (short)0, (c), false, false)

// A-fragment (16x32 bf16, row-major source): lane holds row M = lane&15.
// K run: [half*16 .. half*16+15] (contiguous, matches B fragments)
__device__ __forceinline__ v16bf load_a_frag(const bf16_t* rowBase, int half) {
  v8bf lo = *(const v8bf*)(rowBase + half * 16);
  v8bf hi = *(const v8bf*)(rowBase + half * 16 + 8);
  return __builtin_shufflevector(lo, hi, 0,1,2,3,4,5,6,7,8,9,10,11,12,13,14,15);
}

__device__ __forceinline__ unsigned pack2bf(float a, float b) {
  union { __bf16 h[2]; unsigned u; } u;
  u.h[0] = (bf16_t)a;
  u.h[1] = (bf16_t)b;
  return u.u;
}

// ---------------------------------------------------------------------------
// Prep: convert + swizzle weights into B-fragment order, once per launch.
// Layout: qkv tiles 0..23 then proj tiles 0..7; per tile: 4 k-chunks; per
// chunk: 32 lanes x 16 bf16 (lane n16 = out-col, half = k-half).
// ws element ((jt*4+kc)*32 + lane)*16 + e  =  W[jt*16 + (lane&15)][kc*32 + (lane>>4)*16 + e]
// Total 65536 bf16 = 128 KB in d_ws.
// ---------------------------------------------------------------------------
__global__ __launch_bounds__(256) void swizzle_weights_kernel(
    const float* __restrict__ w_qkv, const float* __restrict__ w_proj,
    bf16_t* __restrict__ wsw) {
  const int i = blockIdx.x * 256 + threadIdx.x;  // 0..65535
  const int i2 = (i < 49152) ? i : i - 49152;
  const float* __restrict__ w = (i < 49152) ? w_qkv : w_proj;
  const int e  = i2 & 15;
  const int ln = (i2 >> 4) & 31;
  const int kc = (i2 >> 9) & 3;
  const int jt = i2 >> 11;
  const int n  = ln & 15;
  const int hf = ln >> 4;
  wsw[i] = (bf16_t)w[(jt * 16 + n) * 128 + kc * 32 + hf * 16 + e];
}

// ---------------------------------------------------------------------------
// Window attention, fused, one block per (batch*window) element.
// x:[4096,49,128] mask:[64,49,49] out:[4096,49,128]
// ---------------------------------------------------------------------------
__global__ __launch_bounds__(256) void swin_window_attn_kernel(
    const float* __restrict__ x, const float* __restrict__ mask,
    const bf16_t* __restrict__ wsw, const float* __restrict__ b_proj,
    float* __restrict__ out) {
  extern __shared__ __align__(128) char smem[];
  bf16_t* qkv = (bf16_t*)smem;              // qs:[4][64][32] | ks:[4][64][32] | vt:[4][32][64]  (48 KB)
  bf16_t* qs = qkv;
  bf16_t* ks = qkv + 8192;
  bf16_t* vt = qkv + 16384;                 // v transposed
  float*  at = (float*)(smem + 49152);      // [4][64][64] f32 logits (64 KB); reused in-place as bf16 probs
  bf16_t* xs = (bf16_t*)(smem + 114688);    // [64][128] bf16: x staging -> mask f32 staging -> attn output
  float*  mf = (float*)(smem + 114688);     // mask window alias [49][49] f32 (9.6 KB of the 16 KB)

  const int b    = blockIdx.x;
  const int tid  = threadIdx.x;
  const int lane = tid & 31;
  const int wave = tid >> 5;                // 0..7
  const int half = lane >> 4;
  const int n16  = lane & 15;

  // ---- Phase 0: stage x -> bf16 LDS (vectorized), zero-pad rows 49..63 ----
#pragma unroll
  for (int it = 0; it < 4; ++it) {
    const int g   = tid + it * 256;         // 0..1023 groups of 8 cols
    const int row = g >> 4;
    const int col = (g & 15) * 8;
    v8bf o;
    if (row < 49) {
      const v4f* px = (const v4f*)(x + (size_t)b * 6272 + row * 128 + col);
      v4f t0 = px[0], t1 = px[1];
#pragma unroll
      for (int q = 0; q < 4; ++q) { o[q] = (bf16_t)t0[q]; o[4 + q] = (bf16_t)t1[q]; }
    } else {
#pragma unroll
      for (int q = 0; q < 8; ++q) o[q] = (bf16_t)0.0f;
    }
    *(v8bf*)(xs + row * 128 + col) = o;
  }
  __syncthreads();

  // ---- Phase 1: qkv = x @ w_qkv^T ----
  // jj = section (0=q,1=k,2=v) at compile time; wave owns col-tile jt = wave + jj*8.
  const int h   = wave >> 1;                // head for this wave's col-tile
  const int hdb = (wave & 1) * 16;          // hd sub-block
#pragma unroll
  for (int jj = 0; jj < 3; ++jj) {
    v16bf bw[4];
    const bf16_t* wb = wsw + (wave + jj * 8) * 2048 + lane * 16;
#pragma unroll
    for (int kc = 0; kc < 4; ++kc) bw[kc] = *(const v16bf*)(wb + kc * 512);
    bf16_t* baseQK = qkv + jj * 8192 + h * 2048 + hdb + n16 + half * 256;  // + (rt*16+r)*32
    bf16_t* baseV  = vt + (h * 32 + hdb + n16) * 64 + half * 8;            // + (rt*16+r)
#pragma unroll
    for (int rt = 0; rt < 4; ++rt) {
      v8f c = {0.f, 0.f, 0.f, 0.f, 0.f, 0.f, 0.f, 0.f};
      const bf16_t* ab = xs + (rt * 16 + n16) * 128;
#pragma unroll
      for (int kc = 0; kc < 4; ++kc) {
        v16bf a = load_a_frag(ab + kc * 32, half);
        c = WMMA_BF16(a, bw[kc], c);
      }
      if (jj == 2) {  // compile-time: v transposed
#pragma unroll
        for (int r = 0; r < 8; ++r) baseV[rt * 16 + r] = (bf16_t)c[r];
      } else {        // q / k row-major
#pragma unroll
        for (int r = 0; r < 8; ++r) baseQK[(rt * 16 + r) * 32] = (bf16_t)c[r];
      }
    }
  }
  __syncthreads();

  // ---- Phase 2: stage mask window into LDS + logits = q @ k^T ----
  for (int i = tid; i < 2401; i += 256)
    mf[i] = mask[(size_t)(b & 63) * 2401 + i];
#pragma unroll
  for (int tt = 0; tt < 8; ++tt) {
    const int t = wave * 8 + tt;
    const int hh = t >> 4, rt = (t >> 2) & 3, kt = t & 3;
    v16bf a = load_a_frag(qs + (hh * 64 + rt * 16 + n16) * 32, half);
    v16bf bb = *(const v16bf*)(ks + (hh * 64 + kt * 16 + n16) * 32 + half * 16);
    v8f c = {0.f, 0.f, 0.f, 0.f, 0.f, 0.f, 0.f, 0.f};
    c = WMMA_BF16(a, bb, c);
#pragma unroll
    for (int r = 0; r < 8; ++r) {
      const int row = rt * 16 + r + half * 8;
      at[(hh * 64 + row) * 64 + kt * 16 + n16] = c[r];
    }
  }
  __syncthreads();

  // ---- Phase 3: softmax, one thread per (head,row); 4*64 = 256 threads ----
  {
    const float scale = 0.17677669529663687f;  // 1/sqrt(32)
    const int hh = tid >> 6, row = tid & 63;
    float* arow = at + (hh * 64 + row) * 64;
    unsigned* prow32 = (unsigned*)arow;  // in-place packed bf16 probabilities
    if (row < 49) {
      v4f lv[13];
#pragma unroll
      for (int j = 0; j < 13; ++j) lv[j] = *(const v4f*)(arow + j * 4);
      const float* mrow = mf + row * 49;
      float l[49];
      float mx = -3.4e38f;
#pragma unroll
      for (int cI = 0; cI < 49; ++cI) {
        l[cI] = lv[cI >> 2][cI & 3] * scale + mrow[cI];
        mx = fmaxf(mx, l[cI]);
      }
      float s = 0.f;
#pragma unroll
      for (int cI = 0; cI < 49; ++cI) {
        float e = __expf(l[cI] - mx);
        l[cI] = e;
        s += e;
      }
      const float inv = 1.0f / s;
#pragma unroll
      for (int i = 0; i < 24; ++i)
        prow32[i] = pack2bf(l[2 * i] * inv, l[2 * i + 1] * inv);
      prow32[24] = pack2bf(l[48] * inv, 0.f);
#pragma unroll
      for (int i = 25; i < 32; ++i) prow32[i] = 0u;
    } else {
#pragma unroll
      for (int i = 0; i < 32; ++i) prow32[i] = 0u;
    }
  }
  __syncthreads();

  // ---- Phase 4: attn_out = P @ V -> xs (reused) as bf16 [64][128] ----
#pragma unroll
  for (int tt = 0; tt < 4; ++tt) {
    const int t = wave * 4 + tt;
    const int hh = t >> 3, rem = t & 7, rt = rem >> 1, ct = rem & 1;
    v8f c = {0.f, 0.f, 0.f, 0.f, 0.f, 0.f, 0.f, 0.f};
    const bf16_t* pbase = (const bf16_t*)(at + (hh * 64 + rt * 16 + n16) * 64);  // row stride 128 bf16
#pragma unroll
    for (int kc = 0; kc < 2; ++kc) {
      v16bf a = load_a_frag(pbase + kc * 32, half);
      v16bf bb = *(const v16bf*)(vt + (hh * 32 + ct * 16 + n16) * 64 + kc * 32 + half * 16);
      c = WMMA_BF16(a, bb, c);
    }
#pragma unroll
    for (int r = 0; r < 8; ++r) {
      const int row = rt * 16 + r + half * 8;
      xs[row * 128 + hh * 32 + ct * 16 + n16] = (bf16_t)c[r];
    }
  }
  __syncthreads();

  // ---- Phase 5: y = o @ w_proj^T + b_proj; f32 stores for rows < 49 ----
  {
    v16bf bw[4];
    const bf16_t* wb = wsw + 49152 + wave * 2048 + lane * 16;
#pragma unroll
    for (int kc = 0; kc < 4; ++kc) bw[kc] = *(const v16bf*)(wb + kc * 512);
    const int jg = wave * 16;
    const float bias = b_proj[jg + n16];
#pragma unroll
    for (int rt = 0; rt < 4; ++rt) {
      v8f c = {0.f, 0.f, 0.f, 0.f, 0.f, 0.f, 0.f, 0.f};
      const bf16_t* ab = xs + (rt * 16 + n16) * 128;
#pragma unroll
      for (int kc = 0; kc < 4; ++kc) {
        v16bf a = load_a_frag(ab + kc * 32, half);
        c = WMMA_BF16(a, bw[kc], c);
      }
#pragma unroll
      for (int r = 0; r < 8; ++r) {
        const int row = rt * 16 + r + half * 8;
        if (row < 49) out[((size_t)b * 49 + row) * 128 + jg + n16] = c[r] + bias;
      }
    }
  }
}

extern "C" void kernel_launch(void* const* d_in, const int* in_sizes, int n_in,
                              void* d_out, int out_size, void* d_ws, size_t ws_size,
                              hipStream_t stream) {
  const float* x      = (const float*)d_in[0];
  const float* mask   = (const float*)d_in[1];
  const float* w_qkv  = (const float*)d_in[2];
  const float* w_proj = (const float*)d_in[3];
  const float* b_proj = (const float*)d_in[4];
  float* out = (float*)d_out;
  bf16_t* wsw = (bf16_t*)d_ws;  // needs 131072 bytes of scratch

  // Prep: swizzle + convert weights (65536 elements, negligible cost)
  swizzle_weights_kernel<<<256, 256, 0, stream>>>(w_qkv, w_proj, wsw);

  const size_t shmem = 131072;  // 128 KB dynamic LDS (2 blocks per 320 KB WGP)
  hipFuncSetAttribute(reinterpret_cast<const void*>(swin_window_attn_kernel),
                      hipFuncAttributeMaxDynamicSharedMemorySize, (int)shmem);
  swin_window_attn_kernel<<<4096, 256, shmem, stream>>>(x, mask, wsw, b_proj, out);
}
